// CustomLSTM_6871947673893
// MI455X (gfx1250) — matrix operand order repaired and checked by reading, not compiled, not run on hardware
//
#include <hip/hip_runtime.h>
#include <hip/hip_bf16.h>

// ---------------------------------------------------------------------------
// 2-layer LSTM for MI455X (gfx1250), bf16 WMMA + fp32 gate math.
//   T=256, B=32, IN=H=1024, gates 4H=4096, PyTorch gate order i,f,g,o.
// Phase A (parallel): G[t,b,:] = x[t,b,:] @ W^T + (bW+bR)      (WMMA bf16)
// Phase B (sequential scan): 16 persistent WGs, each owns 64 hidden units
//   (256 rows of R). Per step: y[t-1] staged to LDS via async loads (GV form),
//   [32 x 256 x 1024] WMMA GEMM (double-buffered), LDS gate exchange,
//   fp32 activations, grid barrier.
// ---------------------------------------------------------------------------

#define T_STEPS 256
#define BATCH   32
#define HID     1024
#define GATES   4096   // 4*HID
#define NWG_REC 16     // persistent workgroups in the recurrent kernel

typedef __bf16 v8bf  __attribute__((ext_vector_type(8)));
typedef __bf16 v16bf __attribute__((ext_vector_type(16)));
typedef float  v8f   __attribute__((ext_vector_type(8)));

// fp32 -> bf16, round-to-nearest-even
__device__ __forceinline__ __bf16 f2bf(float f) {
  unsigned u = __builtin_bit_cast(unsigned, f);
  unsigned r = u + 0x7fffu + ((u >> 16) & 1u);
  unsigned short s = (unsigned short)(r >> 16);
  return __builtin_bit_cast(__bf16, s);
}

__device__ __forceinline__ float sigmoidf_(float x) {
  return 1.0f / (1.0f + __expf(-x));
}

// A fragment: 16x32 bf16, row-major source with row stride lda (elements).
// Lane L<16: M=L, K = k0..k0+7 / k0+16..k0+23; lane L>=16: K bases +8/+24.
template <typename PTR>
__device__ __forceinline__ v16bf load_a_frag(PTR A, int lda, int m0, int k0,
                                             int lane) {
  int row = m0 + (lane & 15);
  int kb  = k0 + ((lane >> 4) << 3);
  auto p = A + (size_t)row * lda + kb;
  v8bf lo = *(const v8bf*)(&p[0]);    // K = kb .. kb+7
  v8bf hi = *(const v8bf*)(&p[16]);   // K = kb+16 .. kb+23
  return __builtin_shufflevector(lo, hi, 0,1,2,3,4,5,6,7,8,9,10,11,12,13,14,15);
}

// B fragment: 32x16 bf16 where B[k][n] = Bt[n][k] (Bt row-major, stride ldb).
// Lane L<16: column N=L, K = k0..k0+7 / k0+16..k0+23; lane L>=16: K bases +8/+24.
__device__ __forceinline__ v16bf load_b_frag(const __bf16* __restrict__ Bt,
                                             int ldb, int n0, int k0, int lane) {
  int col = n0 + (lane & 15);
  int kb  = k0 + ((lane >> 4) << 3);
  const __bf16* p = Bt + (size_t)col * ldb + kb;
  v8bf lo = *(const v8bf*)(p);
  v8bf hi = *(const v8bf*)(p + 16);
  return __builtin_shufflevector(lo, hi, 0,1,2,3,4,5,6,7,8,9,10,11,12,13,14,15);
}

#define WMMA_BF16(A, B, C)                                                    \
  __builtin_amdgcn_wmma_f32_16x16x32_bf16(false, (A), false, (B), (short)0,   \
                                          (C), false, false)

// ------------------------- prep kernels ------------------------------------
__global__ void k_f32_to_bf16(const float* __restrict__ in, __bf16* __restrict__ out, int n) {
  int i = blockIdx.x * 256 + threadIdx.x;
  if (i < n) out[i] = f2bf(in[i]);
}
__global__ void k_copy_f32(const float* __restrict__ in, float* __restrict__ out, int n) {
  int i = blockIdx.x * 256 + threadIdx.x;
  if (i < n) out[i] = in[i];
}
__global__ void k_bias_sum(const float* __restrict__ bW, const float* __restrict__ bR,
                           float* __restrict__ out, int n) {
  int i = blockIdx.x * 256 + threadIdx.x;
  if (i < n) out[i] = bW[i] + bR[i];
}
__global__ void k_zero_i32(int* __restrict__ p, int n) {
  int i = blockIdx.x * 256 + threadIdx.x;
  if (i < n) p[i] = 0;
}

// ------------------------- input GEMM --------------------------------------
// C[M,N] = A[M,K] * Bt[N,K]^T + bias[N].
// Block tile 32M x 256N; 8 waves, each 2Mx2N tiles (4 WMMA / K-step),
// software-pipelined (double-buffered fragments).
__global__ __launch_bounds__(256)
void gemm_bias_kernel(const __bf16* __restrict__ A, const __bf16* __restrict__ Bt,
                      const float* __restrict__ bias, float* __restrict__ C,
                      int M, int N, int K) {
  const int w    = threadIdx.x >> 5;
  const int lane = threadIdx.x & 31;
  const int n0   = blockIdx.x * 256 + w * 32;   // two N tiles: n0, n0+16
  const int m0   = blockIdx.y * 32;             // two M tiles: m0, m0+16

  v8f acc00 = {}, acc01 = {}, acc10 = {}, acc11 = {};

  v16bf b0c = load_b_frag(Bt, K, n0,      0, lane);
  v16bf b1c = load_b_frag(Bt, K, n0 + 16, 0, lane);
  v16bf a0c = load_a_frag(A, K, m0,      0, lane);
  v16bf a1c = load_a_frag(A, K, m0 + 16, 0, lane);

  for (int k0 = 0; k0 < K; k0 += 32) {
    const int kn = (k0 + 32 < K) ? (k0 + 32) : 0;   // clamped: unconditional
    v16bf b0n = load_b_frag(Bt, K, n0,      kn, lane);
    v16bf b1n = load_b_frag(Bt, K, n0 + 16, kn, lane);
    v16bf a0n = load_a_frag(A, K, m0,      kn, lane);
    v16bf a1n = load_a_frag(A, K, m0 + 16, kn, lane);
    if (k0 + 96 < K)
      __builtin_prefetch(Bt + (size_t)(n0 + lane) * K + k0 + 96, 0, 1);

    acc00 = WMMA_BF16(a0c, b0c, acc00);
    acc01 = WMMA_BF16(a0c, b1c, acc01);
    acc10 = WMMA_BF16(a1c, b0c, acc10);
    acc11 = WMMA_BF16(a1c, b1c, acc11);

    b0c = b0n; b1c = b1n; a0c = a0n; a1c = a1n;
  }

  const int colr   = lane & 15;
  const int rowsel = (lane >> 4) * 8;
  const int na = n0 + colr;
  const int nb = n0 + 16 + colr;
  const float ba = bias[na];
  const float bb = bias[nb];
#pragma unroll
  for (int r = 0; r < 8; ++r) {
    size_t rowA = (size_t)(m0 + rowsel + r) * N;
    size_t rowB = (size_t)(m0 + 16 + rowsel + r) * N;
    C[rowA + na] = acc00[r] + ba;
    C[rowA + nb] = acc01[r] + bb;
    C[rowB + na] = acc10[r] + ba;
    C[rowB + nb] = acc11[r] + bb;
  }
}

// ------------------------- recurrent scan ----------------------------------
// Persistent kernel, NWG_REC workgroups; WG b owns hidden units [64b, 64b+64).
// LDS: y[t-1] stage (64KB, filled via global_load_async_to_lds_b128, GV form)
//      + gate exchange buffer (32KB).
__global__ __launch_bounds__(256)
void lstm_recurrent_kernel(const float*  __restrict__ G,      // [T][B][4H]
                           const __bf16* __restrict__ Rbf,    // [4H][H]
                           const __bf16* __restrict__ yinit,  // [B][H]
                           __bf16*       __restrict__ ybf,    // [T][B][H]
                           float*        __restrict__ cstate, // [B][H]
                           float*        __restrict__ yout,   // [T][B][H] or null
                           float*        __restrict__ hn,     // [B][H]
                           float*        __restrict__ cn,     // [B][H]
                           int*          __restrict__ bar_cnt,
                           int T) {
  __shared__ __align__(16) __bf16 ysh[BATCH][HID];   // 64 KB
  __shared__ float gbuf[BATCH][256];                 // 32 KB: 4 gates x 64 h

  const int w    = threadIdx.x >> 5;
  const int lane = threadIdx.x & 31;
  const int h0   = blockIdx.x * 64;
  const int gate = w >> 1;
  // two N tiles per wave (adjacent 16-row groups within one gate)
  const int rowA = gate * HID + h0 + (w & 1) * 32;
  const int rowB = rowA + 16;

  const unsigned ysh_base = (unsigned)(size_t)(&ysh[0][0]);  // LDS byte offset

  for (int t = 0; t < T; ++t) {
    const __bf16* Aprev =
        (t == 0) ? yinit : (ybf + (size_t)(t - 1) * BATCH * HID);

    // ---- stage y[t-1] into LDS with CDNA5 async loads (ASYNCcnt) ----
    // GV form: 64-bit address in VGPR pair, LDS byte offset in VGPR.
    // 32*1024*2 = 65536 B = 4096 x b128; 16 chunks per thread.
#pragma unroll
    for (int i = 0; i < 16; ++i) {
      unsigned idx  = (unsigned)threadIdx.x + (unsigned)i * 256u;
      unsigned loff = ysh_base + idx * 16u;
      unsigned long long gaddr =
          (unsigned long long)(size_t)Aprev + (unsigned long long)idx * 16ull;
      asm volatile("global_load_async_to_lds_b128 %0, %1, off"
                   :: "v"(loff), "v"(gaddr) : "memory");
    }
    asm volatile("s_wait_asynccnt 0x0" ::: "memory");
    __syncthreads();

    // ---- GEMM: gates[h-chunk] = y[t-1] @ R_chunk^T (double-buffered) ----
    v8f acc00 = {}, acc01 = {}, acc10 = {}, acc11 = {};
    const __bf16* ysh_p = &ysh[0][0];

    v16bf b0c = load_b_frag(Rbf, HID, rowA, 0, lane);
    v16bf b1c = load_b_frag(Rbf, HID, rowB, 0, lane);
    v16bf a0c = load_a_frag(ysh_p, HID, 0,  0, lane);
    v16bf a1c = load_a_frag(ysh_p, HID, 16, 0, lane);

    for (int k0 = 0; k0 < HID; k0 += 32) {
      const int kn = (k0 + 32 < HID) ? (k0 + 32) : 0;
      v16bf b0n = load_b_frag(Rbf, HID, rowA, kn, lane);
      v16bf b1n = load_b_frag(Rbf, HID, rowB, kn, lane);
      v16bf a0n = load_a_frag(ysh_p, HID, 0,  kn, lane);
      v16bf a1n = load_a_frag(ysh_p, HID, 16, kn, lane);

      acc00 = WMMA_BF16(a0c, b0c, acc00);
      acc01 = WMMA_BF16(a0c, b1c, acc01);
      acc10 = WMMA_BF16(a1c, b0c, acc10);
      acc11 = WMMA_BF16(a1c, b1c, acc11);

      b0c = b0n; b1c = b1n; a0c = a0n; a1c = a1n;
    }

    // ---- gates += input-GEMM contribution; exchange via LDS ----
    const int colr   = lane & 15;
    const int rowsel = (lane >> 4) * 8;
    const int nlocA  = (2 * w) * 16 + colr;          // 0..255 within WG
    const int nlocB  = nlocA + 16;
    const int growA  = rowA + colr;                  // global gate indices
    const int growB  = rowB + colr;
    const float* Gt  = G + (size_t)t * BATCH * GATES;
#pragma unroll
    for (int r = 0; r < 8; ++r) {
      int bA = rowsel + r;        // batch rows for M-tile 0
      int bB = bA + 16;           // batch rows for M-tile 1
      gbuf[bA][nlocA] = acc00[r] + Gt[(size_t)bA * GATES + growA];
      gbuf[bA][nlocB] = acc01[r] + Gt[(size_t)bA * GATES + growB];
      gbuf[bB][nlocA] = acc10[r] + Gt[(size_t)bB * GATES + growA];
      gbuf[bB][nlocB] = acc11[r] + Gt[(size_t)bB * GATES + growB];
    }
    __syncthreads();

    // ---- elementwise LSTM cell: 2048 (b,h) pairs / 256 threads ----
#pragma unroll
    for (int j = 0; j < 8; ++j) {
      int idx = threadIdx.x + j * 256;
      int b = idx >> 6;
      int h = idx & 63;
      float iv = sigmoidf_(gbuf[b][h]);
      float fv = sigmoidf_(gbuf[b][64 + h]);
      float gv = tanhf(gbuf[b][128 + h]);
      float ov = sigmoidf_(gbuf[b][192 + h]);
      size_t ci = (size_t)b * HID + h0 + h;
      float c = fv * cstate[ci] + iv * gv;
      cstate[ci] = c;
      float y = ov * tanhf(c);
      size_t yi = (size_t)t * BATCH * HID + ci;
      ybf[yi] = f2bf(y);
      if (yout) yout[yi] = y;
      if (t == T - 1) { hn[ci] = y; cn[ci] = c; }
    }
    __syncthreads();

    // ---- grid barrier: y[t] globally visible before step t+1 ----
    if (threadIdx.x == 0) {
      __threadfence();
      atomicAdd(bar_cnt, 1);
      const int target = NWG_REC * (t + 1);
      while (__hip_atomic_load(bar_cnt, __ATOMIC_ACQUIRE,
                               __HIP_MEMORY_SCOPE_AGENT) < target) {
        __builtin_amdgcn_s_sleep(1);
      }
    }
    __syncthreads();
  }
}

// ------------------------- host orchestration ------------------------------
extern "C" void kernel_launch(void* const* d_in, const int* in_sizes, int n_in,
                              void* d_out, int out_size, void* d_ws, size_t ws_size,
                              hipStream_t stream) {
  (void)in_sizes; (void)n_in; (void)out_size; (void)ws_size;

  const float* x   = (const float*)d_in[0];   // [T,B,IN]
  const float* h0  = (const float*)d_in[1];   // [2,B,H]
  const float* c0  = (const float*)d_in[2];   // [2,B,H]
  const float* W0  = (const float*)d_in[3];   // [4H,IN]
  const float* R0  = (const float*)d_in[4];   // [4H,H]
  const float* bW0 = (const float*)d_in[5];
  const float* bR0 = (const float*)d_in[6];
  const float* W1  = (const float*)d_in[7];   // [4H,H]
  const float* R1  = (const float*)d_in[8];
  const float* bW1 = (const float*)d_in[9];
  const float* bR1 = (const float*)d_in[10];

  const int TBH = T_STEPS * BATCH * HID;   // 8388608
  const int BH  = BATCH * HID;             // 32768

  float* y2_out = (float*)d_out;                 // [T,B,H]
  float* hn0 = (float*)d_out + TBH;              // h_n layer0
  float* hn1 = (float*)d_out + TBH + BH;         // h_n layer1
  float* cn0 = (float*)d_out + TBH + 2 * BH;     // c_n layer0
  float* cn1 = (float*)d_out + TBH + 3 * BH;     // c_n layer1

  // workspace carve-up (256B aligned)
  char* ws = (char*)d_ws;
  size_t off = 0;
  auto carve = [&](size_t bytes) -> char* {
    char* p = ws + off;
    off += (bytes + 255) & ~(size_t)255;
    return p;
  };
  __bf16* Wbf0  = (__bf16*)carve((size_t)GATES * HID * 2);   // 8 MB
  __bf16* Rbf0  = (__bf16*)carve((size_t)GATES * HID * 2);   // 8 MB
  __bf16* Wbf1  = (__bf16*)carve((size_t)GATES * HID * 2);   // 8 MB
  __bf16* Rbf1  = (__bf16*)carve((size_t)GATES * HID * 2);   // 8 MB
  __bf16* xbf   = (__bf16*)carve((size_t)TBH * 2);           // 16 MB (reused as y2bf)
  __bf16* y1bf  = (__bf16*)carve((size_t)TBH * 2);           // 16 MB
  float*  Gbuf  = (float*) carve((size_t)T_STEPS * BATCH * GATES * 4); // 128 MB
  float*  b0    = (float*) carve((size_t)GATES * 4);
  float*  b1    = (float*) carve((size_t)GATES * 4);
  __bf16* yini0 = (__bf16*)carve((size_t)BH * 2);
  __bf16* yini1 = (__bf16*)carve((size_t)BH * 2);
  float*  cst0  = (float*) carve((size_t)BH * 4);
  float*  cst1  = (float*) carve((size_t)BH * 4);
  int*    cnts  = (int*)   carve(256);
  __bf16* y2bf  = xbf;   // layer-0 input GEMM done before layer-1 scan -> safe reuse

  auto nblk = [](int n) { return (n + 255) / 256; };
  const int WN = GATES * HID;   // 4194304 weight elements

  // --- prep ---
  k_zero_i32   <<<1, 256, 0, stream>>>(cnts, 2);
  k_bias_sum   <<<nblk(GATES), 256, 0, stream>>>(bW0, bR0, b0, GATES);
  k_bias_sum   <<<nblk(GATES), 256, 0, stream>>>(bW1, bR1, b1, GATES);
  k_f32_to_bf16<<<nblk(WN),  256, 0, stream>>>(W0, Wbf0, WN);
  k_f32_to_bf16<<<nblk(WN),  256, 0, stream>>>(R0, Rbf0, WN);
  k_f32_to_bf16<<<nblk(WN),  256, 0, stream>>>(W1, Wbf1, WN);
  k_f32_to_bf16<<<nblk(WN),  256, 0, stream>>>(R1, Rbf1, WN);
  k_f32_to_bf16<<<nblk(TBH), 256, 0, stream>>>(x, xbf, TBH);
  k_f32_to_bf16<<<nblk(BH),  256, 0, stream>>>(h0,      yini0, BH);
  k_f32_to_bf16<<<nblk(BH),  256, 0, stream>>>(h0 + BH, yini1, BH);
  k_copy_f32   <<<nblk(BH),  256, 0, stream>>>(c0,      cst0, BH);
  k_copy_f32   <<<nblk(BH),  256, 0, stream>>>(c0 + BH, cst1, BH);

  const dim3 ggrid(GATES / 256, (T_STEPS * BATCH) / 32);  // (16, 256)

  // --- layer 0 ---
  gemm_bias_kernel<<<ggrid, 256, 0, stream>>>(xbf, Wbf0, b0, Gbuf,
                                              T_STEPS * BATCH, GATES, HID);
  lstm_recurrent_kernel<<<NWG_REC, 256, 0, stream>>>(
      Gbuf, Rbf0, yini0, y1bf, cst0, /*yout=*/nullptr, hn0, cn0,
      cnts + 0, T_STEPS);

  // --- layer 1 ---
  gemm_bias_kernel<<<ggrid, 256, 0, stream>>>(y1bf, Wbf1, b1, Gbuf,
                                              T_STEPS * BATCH, GATES, HID);
  lstm_recurrent_kernel<<<NWG_REC, 256, 0, stream>>>(
      Gbuf, Rbf1, yini1, y2bf, cst1, /*yout=*/y2_out, hn1, cn1,
      cnts + 1, T_STEPS);
}
